// MySelfAttention_72988674228413
// MI455X (gfx1250) — matrix operand order repaired and checked by reading, not compiled, not run on hardware
//
#include <hip/hip_runtime.h>
#include <hip/hip_bf16.h>

typedef __attribute__((ext_vector_type(16))) __bf16 v16bf;
typedef __attribute__((ext_vector_type(8)))  __bf16 v8bf;
typedef __attribute__((ext_vector_type(8)))  float  v8f;
typedef __attribute__((ext_vector_type(4)))  float  f32x4;
typedef __attribute__((ext_vector_type(4)))  int    v4i;

namespace {
constexpr int kB = 4;
constexpr int kS = 4096;
constexpr int kD = 256;
constexpr int kWaves = 4;                              // waves per workgroup
constexpr int kRowsPerWave = 16;                       // WMMA M tile
constexpr int kRowsPerBlock = kWaves * kRowsPerWave;   // 64
constexpr int kKeyTile = 32;                           // keys per iteration
constexpr int kKPad = kD + 8;    // 264: 528B rows -> 16B aligned, conflict-free
constexpr int kVPad = kKeyTile + 8;  // 40: 80B rows -> 16B aligned, conflict-free
}

#if __has_builtin(__builtin_amdgcn_global_load_async_to_lds_b128)
#define HAS_ASYNC_LDS 1
#else
#define HAS_ASYNC_LDS 0
#endif

__device__ __forceinline__ void wait_async0() {
#if HAS_ASYNC_LDS
#if __has_builtin(__builtin_amdgcn_s_wait_asynccnt)
  __builtin_amdgcn_s_wait_asynccnt(0);
#else
  asm volatile("s_wait_asynccnt 0x0" ::: "memory");
#endif
#endif
}

// Copy 16 bytes global -> LDS. Async (ASYNCcnt / global_load_async_to_lds_b128)
// when available; signature is (v4i AS1* src, v4i AS3* dst, imm offset, imm cpol).
__device__ __forceinline__ void copy16_g2l(__bf16* lds, const __bf16* g) {
#if HAS_ASYNC_LDS
  __builtin_amdgcn_global_load_async_to_lds_b128(
      (__attribute__((address_space(1))) v4i*)g,
      (__attribute__((address_space(3))) v4i*)lds, 0, 0);
#else
  *(v8bf*)lds = *(const v8bf*)g;
#endif
}

__device__ __forceinline__ v8f wmma_bf16(v16bf a, v16bf b, v8f c) {
  // D = A(16x32 bf16) * B(32x16 bf16) + C(16x16 f32)
  return __builtin_amdgcn_wmma_f32_16x16x32_bf16(false, a, false, b, (short)0, c,
                                                 false, false);
}

__device__ __forceinline__ v8f zero8() {
  v8f z;
#pragma unroll
  for (int i = 0; i < 8; ++i) z[i] = 0.0f;
  return z;
}

// A fragment (16x32, 16-bit) from a row-major bf16 matrix, row stride = ld.
// Lane L: row = L&15; elements 0..7 -> K = 8*hi + 0..7, 8..15 -> K = 8*hi+16..23.
__device__ __forceinline__ v16bf load_a_bf16(const __bf16* base, int ld, int lane) {
  const int row = lane & 15, hi = lane >> 4;
  const __bf16* p = base + (size_t)row * ld + hi * 8;
  v8bf a = *(const v8bf*)(p);
  v8bf b = *(const v8bf*)(p + 16);
  v16bf r;
#pragma unroll
  for (int i = 0; i < 8; ++i) { r[i] = a[i]; r[i + 8] = b[i]; }
  return r;
}

// A fragment from row-major fp32, converted to bf16 on the fly.
__device__ __forceinline__ v16bf load_a_f32(const float* base, int ld, int lane) {
  const int row = lane & 15, hi = lane >> 4;
  const float* p = base + (size_t)row * ld + hi * 8;
  f32x4 x0 = *(const f32x4*)(p);
  f32x4 x1 = *(const f32x4*)(p + 4);
  f32x4 x2 = *(const f32x4*)(p + 16);
  f32x4 x3 = *(const f32x4*)(p + 20);
  v16bf r;
#pragma unroll
  for (int i = 0; i < 4; ++i) {
    r[i]      = (__bf16)x0[i];
    r[i + 4]  = (__bf16)x1[i];
    r[i + 8]  = (__bf16)x2[i];
    r[i + 12] = (__bf16)x3[i];
  }
  return r;
}

// B fragment (32x16, 16-bit), column-major semantics. Source stored so that each
// B-column is a contiguous run of >=32 bf16, row stride = ld.
// Lane L: col = L&15; elements 0..7 -> K = 8*hi + 0..7, 8..15 -> K = 8*hi+16..23.
__device__ __forceinline__ v16bf load_b_bf16(const __bf16* base, int ld, int lane) {
  const int n = lane & 15, hi = lane >> 4;
  const __bf16* p = base + (size_t)n * ld + hi * 8;
  v8bf a = *(const v8bf*)(p);
  v8bf b = *(const v8bf*)(p + 16);
  v16bf r;
#pragma unroll
  for (int i = 0; i < 8; ++i) { r[i] = a[i]; r[i + 8] = b[i]; }
  return r;
}

// -------- Stage 0: W[din][dout] fp32 -> Wt[m][dout][din] bf16 --------
__global__ void prep_weights_kernel(const float* __restrict__ Wq,
                                    const float* __restrict__ Wk,
                                    const float* __restrict__ Wv,
                                    __bf16* __restrict__ wt) {
  int idx = blockIdx.x * blockDim.x + threadIdx.x;
  if (idx >= 3 * kD * kD) return;
  const int m = idx / (kD * kD);
  const int r = idx % (kD * kD);
  const int nout = r / kD;
  const int kin = r % kD;
  const float* W = (m == 0) ? Wq : (m == 1) ? Wk : Wv;
  wt[idx] = (__bf16)W[kin * kD + nout];
}

// -------- Stage 1: Q/K/V projections (bf16 WMMA, f32 accumulate) --------
__global__ __launch_bounds__(kWaves * 32, 1) void qkv_proj_kernel(
    const float* __restrict__ x, const float* __restrict__ bq,
    const float* __restrict__ bk, const float* __restrict__ bv,
    const __bf16* __restrict__ wt, __bf16* __restrict__ qbf,
    __bf16* __restrict__ kbf, __bf16* __restrict__ vt) {
  const int lane = threadIdx.x & 31;
  const int wave = threadIdx.x >> 5;
  const int rowbase = (blockIdx.x * kWaves + wave) * kRowsPerWave;
  const int n = lane & 15;
  const int hi = lane >> 4;

  v16bf af[8];  // x rows [rowbase, rowbase+16), K = 256 as 8 chunks of 32
#pragma unroll
  for (int c = 0; c < 8; ++c)
    af[c] = load_a_f32(x + (size_t)rowbase * kD + c * 32, kD, lane);

  const float* biases[3] = {bq, bk, bv};
#pragma unroll
  for (int m = 0; m < 3; ++m) {
    const __bf16* wtm = wt + (size_t)m * kD * kD;
    for (int nt = 0; nt < kD / 16; ++nt) {
      v8f acc = zero8();
#pragma unroll
      for (int c = 0; c < 8; ++c) {
        v16bf bfrag = load_b_bf16(wtm + (size_t)(nt * 16) * kD + c * 32, kD, lane);
        acc = wmma_bf16(af[c], bfrag, acc);
      }
      const int col = nt * 16 + n;
      const float bias = biases[m][col];  // per-column bias, uniform over rows
#pragma unroll
      for (int i = 0; i < 8; ++i) acc[i] += bias;

      if (m == 2) {
        // V stored transposed: Vt[b][d][t] (t contiguous over element index)
        const int b = rowbase >> 12;                 // rowbase / kS
        const int t0 = (rowbase & (kS - 1)) + hi * 8;
        v8bf vvec;
#pragma unroll
        for (int i = 0; i < 8; ++i) vvec[i] = (__bf16)acc[i];
        *(v8bf*)(vt + ((size_t)b * kD + col) * kS + t0) = vvec;
      } else {
        __bf16* dst = (m == 0) ? qbf : kbf;
#pragma unroll
        for (int i = 0; i < 8; ++i)
          dst[(size_t)(rowbase + i + 8 * hi) * kD + col] = (__bf16)acc[i];
      }
    }
  }
}

// -------- Stage 2: flash attention, LDS-shared K/V, async double buffer ------
__global__ __launch_bounds__(kWaves * 32, 1) void attn_kernel(
    const __bf16* __restrict__ qbf, const __bf16* __restrict__ kbf,
    const __bf16* __restrict__ vt, float* __restrict__ out) {
  __shared__ __bf16 kbuf[2][kKeyTile][kKPad];  // 2*32*264*2 = 33792 B
  __shared__ __bf16 vbuf[2][kD][kVPad];        // 2*256*40*2 = 40960 B
  __shared__ __bf16 pbuf[kWaves][16][32];      // per-wave P transpose bounce

  const int tid = threadIdx.x;
  const int lane = tid & 31;
  const int wave = tid >> 5;
  const int rowbase = (blockIdx.x * kWaves + wave) * kRowsPerWave;
  const int b = rowbase >> 12;  // rowbase / kS (tiles never cross batches)
  const int n = lane & 15;
  const int hi = lane >> 4;

  const __bf16* qb = qbf + (size_t)rowbase * kD;
  const __bf16* kb = kbf + (size_t)b * kS * kD;
  const __bf16* vb = vt + (size_t)b * kD * kS;

  // Cooperative staging of one 32-key K tile + V tile into LDS (16B packets).
  auto stage = [&](int key0, int buf) {
    // K tile: 32 rows x 256 cols = 1024 packets of 8 bf16; 8 per thread
#pragma unroll
    for (int j = 0; j < 8; ++j) {
      const int idx = tid + 128 * j;
      const int row = idx >> 5, col = (idx & 31) * 8;
      copy16_g2l(&kbuf[buf][row][col], kb + (size_t)(key0 + row) * kD + col);
    }
    // V tile: 256 d-rows x 32 keys = 1024 packets of 8 bf16; 8 per thread
#pragma unroll
    for (int j = 0; j < 8; ++j) {
      const int idx = tid + 128 * j;
      const int d = idx >> 2, col = (idx & 3) * 8;
      copy16_g2l(&vbuf[buf][d][col], vb + (size_t)d * kS + key0 + col);
    }
  };

  v16bf qf[8];  // Q tile resident in registers (16 x 256)
#pragma unroll
  for (int c = 0; c < 8; ++c) qf[c] = load_a_bf16(qb + c * 32, kD, lane);

  v8f o[16];  // O accumulator 16 x 256 f32
#pragma unroll
  for (int nt = 0; nt < 16; ++nt) o[nt] = zero8();

  float mrow[8], lrow[8];  // element i <-> query row (i + 8*hi)
#pragma unroll
  for (int i = 0; i < 8; ++i) { mrow[i] = -3.0e38f; lrow[i] = 0.0f; }

  const float scale = 0.0625f;  // 1/sqrt(256)
  constexpr int kTiles = kS / kKeyTile;  // 128

  stage(0, 0);
  wait_async0();
  __syncthreads();

  for (int kt = 0; kt < kTiles; ++kt) {
    const int buf = kt & 1;
    if (kt + 1 < kTiles) stage((kt + 1) * kKeyTile, buf ^ 1);  // prefetch next

    // S tile = Q * K^T for 32 keys (two 16-column WMMA tiles), from LDS
    v8f s0 = zero8(), s1 = zero8();
#pragma unroll
    for (int c = 0; c < 8; ++c) {
      v16bf k0 = load_b_bf16(&kbuf[buf][0][c * 32], kKPad, lane);
      v16bf k1 = load_b_bf16(&kbuf[buf][16][c * 32], kKPad, lane);
      s0 = wmma_bf16(qf[c], k0, s0);
      s1 = wmma_bf16(qf[c], k1, s1);
    }
#pragma unroll
    for (int i = 0; i < 8; ++i) { s0[i] *= scale; s1[i] *= scale; }

    // Online softmax. Row reductions: xor-shuffles with masks 1/2/4/8 stay
    // inside each 16-lane half (which holds all 16 columns of rows 0-7 / 8-15).
    float p0[8], p1[8], alpha[8];
#pragma unroll
    for (int i = 0; i < 8; ++i) {
      float t = fmaxf(s0[i], s1[i]);
#pragma unroll
      for (int mask = 1; mask < 16; mask <<= 1)
        t = fmaxf(t, __shfl_xor(t, mask, 32));
      const float mnew = fmaxf(mrow[i], t);
      alpha[i] = __expf(mrow[i] - mnew);
      mrow[i] = mnew;
      p0[i] = __expf(s0[i] - mnew);
      p1[i] = __expf(s1[i] - mnew);
      float rs = p0[i] + p1[i];
#pragma unroll
      for (int mask = 1; mask < 16; mask <<= 1)
        rs += __shfl_xor(rs, mask, 32);
      lrow[i] = lrow[i] * alpha[i] + rs;
    }
#pragma unroll
    for (int nt = 0; nt < 16; ++nt)
#pragma unroll
      for (int i = 0; i < 8; ++i) o[nt][i] *= alpha[i];

    // C/D layout -> A layout transpose through per-wave LDS (wave-synchronous).
#pragma unroll
    for (int i = 0; i < 8; ++i) {
      pbuf[wave][i + 8 * hi][n] = (__bf16)p0[i];
      pbuf[wave][i + 8 * hi][n + 16] = (__bf16)p1[i];
    }
    v16bf pf = load_a_bf16(&pbuf[wave][0][0], 32, lane);

    // O += P * V from LDS (V rows are d-columns; keys contiguous)
#pragma unroll
    for (int nt = 0; nt < 16; ++nt) {
      v16bf vf = load_b_bf16(&vbuf[buf][nt * 16][0], kVPad, lane);
      o[nt] = wmma_bf16(pf, vf, o[nt]);
    }

    wait_async0();    // our share of next tile's async copies landed
    __syncthreads();  // everyone's copies landed; safe to flip buffers
  }

#pragma unroll
  for (int i = 0; i < 8; ++i) lrow[i] = 1.0f / lrow[i];
#pragma unroll
  for (int nt = 0; nt < 16; ++nt)
#pragma unroll
    for (int i = 0; i < 8; ++i)
      out[(size_t)(rowbase + i + 8 * hi) * kD + nt * 16 + n] = o[nt][i] * lrow[i];
}

extern "C" void kernel_launch(void* const* d_in, const int* in_sizes, int n_in,
                              void* d_out, int out_size, void* d_ws, size_t ws_size,
                              hipStream_t stream) {
  const float* x = (const float*)d_in[0];
  const float* Wq = (const float*)d_in[1];
  const float* bq = (const float*)d_in[2];
  const float* Wk = (const float*)d_in[3];
  const float* bk = (const float*)d_in[4];
  const float* Wv = (const float*)d_in[5];
  const float* bv = (const float*)d_in[6];
  float* out = (float*)d_out;

  // Workspace layout (bf16 elements): Wt[3][D][D] | Q[B*S][D] | K[B*S][D] | Vt[B][D][S]
  __bf16* ws = (__bf16*)d_ws;
  __bf16* wt = ws;
  __bf16* qbf = wt + (size_t)3 * kD * kD;
  __bf16* kbf = qbf + (size_t)kB * kS * kD;
  __bf16* vt = kbf + (size_t)kB * kS * kD;

  {
    const int nw = 3 * kD * kD;
    prep_weights_kernel<<<(nw + 255) / 256, 256, 0, stream>>>(Wq, Wk, Wv, wt);
  }
  const int nblocks = (kB * kS) / kRowsPerBlock;  // 256
  qkv_proj_kernel<<<nblocks, kWaves * 32, 0, stream>>>(x, bq, bk, bv, wt, qbf, kbf, vt);
  attn_kernel<<<nblocks, kWaves * 32, 0, stream>>>(qbf, kbf, vt, out);
}
